// vSU2_CNN_triple_41979010351325
// MI455X (gfx1250) — matrix operand order repaired and checked by reading, not compiled, not run on hardware
//
#include <hip/hip_runtime.h>

// ---------------------------------------------------------------------------
// vSU2 CNN triple — CDNA5 (gfx1250) WMMA + Tensor-Data-Mover implementation
//
//   1) prep_weights : pack W0 (1024x128), W1/W2 (4096x128) f16, B-fragment order
//   2) layer0       : P tile (16x1024 f16) built in LDS from x + TRIPLE,
//                     v_wmma_f32_16x16x32_f16 vs W0, softplus -> y0 (f16,
//                     rows 0..62 replicated at tail for wrap-free staging)
//   3) conv x2      : TDM tensor_load_to_lds stages 63 contiguous rows
//                     (63 x 256B, +16B LDS pad/row via pad_enable) in one DMA;
//                     8 waves x 2 M-tiles x 128 K-chunks of WMMA; softplus
//   4) reduce       : (1/N) * sum_i y2[i,:]·dense_w + dense_b -> scalar f32
//
// Fragment layout assumptions (CDNA5 ISA §7.12.2, wave32):
//   A 16x32 f16 : lane l: M=l%16; K = {B..B+7} u {B+16..B+23}, B = 8*(l/16)
//   B 32x16 f16 : lane l: N=l%16; K = {B..B+7} u {B+16..B+23}, B = 8*(l/16)
//   C/D 16x16 f32: lane l: N=l%16; VGPR v: M = v + 8*(l/16)
// ---------------------------------------------------------------------------

typedef _Float16 f16;
typedef __attribute__((ext_vector_type(8)))  _Float16 v8h;
typedef __attribute__((ext_vector_type(16))) _Float16 v16h;
typedef __attribute__((ext_vector_type(8)))  float    v8f;
typedef __attribute__((ext_vector_type(4)))  unsigned int u32x4;
typedef __attribute__((ext_vector_type(8)))  int      i32x8;
typedef __attribute__((ext_vector_type(4)))  int      i32x4;

#define NROWS 32768
#define NMASK (NROWS - 1)
#define YPAD  64                 // replicated tail rows (need 63)

static __device__ __forceinline__ float softplus_f(float v) {
    // matches jax.nn.softplus: max(v,0) + log1p(exp(-|v|))
    return fmaxf(v, 0.f) + log1pf(expf(-fabsf(v)));
}

static __device__ __forceinline__ v16h join8(v8h lo, v8h hi) {
    union { v16h v; v8h h[2]; } u;
    u.h[0] = lo; u.h[1] = hi;
    return u.v;
}

// ---------------------------------------------------------------------------
// Pack weights into f16 B-fragment order.
// p = ((kc*8 + ft)*32 + l)*16 + e ; K = kc*32 + (e<8 ? 8*(l>>4)+e : 16+8*(l>>4)+(e-8)), F = ft*16+(l&15)
// ---------------------------------------------------------------------------
__global__ void prep_weights(const float* __restrict__ ik,
                             const float* __restrict__ k0,
                             const float* __restrict__ k1,
                             f16* __restrict__ W0, f16* __restrict__ W1,
                             f16* __restrict__ W2) {
    int p = blockIdx.x * blockDim.x + threadIdx.x;
    int layer, q;
    if (p < 131072)               { layer = 0; q = p; }
    else if (p < 131072 + 524288) { layer = 1; q = p - 131072; }
    else if (p < 131072 + 1048576){ layer = 2; q = p - 131072 - 524288; }
    else return;

    int tile = q >> 9, rem = q & 511;
    int l = rem >> 4, e = rem & 15;
    int kc = tile >> 3, ft = tile & 7;
    int Kl = (e < 8) ? (8 * (l >> 4) + e) : (16 + 8 * (l >> 4) + (e - 8));
    int Kg = kc * 32 + Kl;
    int F  = ft * 16 + (l & 15);

    if (layer == 0) {
        int j = Kg >> 5, k = Kg & 31;
        float val = ik[F * 1024 + j * 32 + k] - ik[F * 1024 + k * 32 + j];
        W0[q] = (f16)val;
    } else {
        int m = Kg >> 7, c = Kg & 127;
        const float* src = (layer == 1) ? k0 : k1;
        ((layer == 1) ? W1 : W2)[q] = (f16)src[(m * 128 + c) * 128 + F];
    }
}

// ---------------------------------------------------------------------------
// Layer 0: y0[i,f] = softplus( sum_{j,k} A[f,j,k] * T[x_i, x_{i+1+j}, x_{i+1+k}] )
// ---------------------------------------------------------------------------
__global__ __launch_bounds__(256) void layer0_kernel(const int* __restrict__ x,
                                                     const f16* __restrict__ W0,
                                                     f16* __restrict__ y0) {
    __shared__ float trip[64];
    __shared__ int   xw[48];
    __shared__ f16   P[16 * 1032];

    const int tid = threadIdx.x;
    const int i0  = blockIdx.x * 16;

    if (tid < 64) {
        int a = tid >> 4, b = (tid >> 2) & 3, c = tid & 3;
        float v = 0.f;
        if (a != b && a != c && b != c) {
            int s = (b - a) * (c - a) * (c - b);
            float base = ((a + b + c) & 1) ? 0.769800358919501f
                                           : -0.769800358919501f;
            v = (s > 0) ? base : -base;
        }
        trip[tid] = v;
    }
    if (tid < 48) xw[tid] = x[(i0 + tid) & NMASK];
    __syncthreads();

    for (int idx = tid; idx < 16 * 1024; idx += 256) {
        int ii = idx >> 10, jk = idx & 1023;
        int j = jk >> 5, k = jk & 31;
        P[ii * 1032 + jk] =
            (f16)trip[(xw[ii] << 4) + (xw[ii + 1 + j] << 2) + xw[ii + 1 + k]];
    }
    __syncthreads();

    const int wave = tid >> 5, lane = tid & 31;
    const int ii = lane & 15, hf = lane >> 4;

    v8f acc = {};
#pragma unroll 4
    for (int kc = 0; kc < 32; ++kc) {
        const f16* ap = &P[ii * 1032 + kc * 32 + hf * 8];
        v16h a = join8(*(const v8h*)ap, *(const v8h*)(ap + 16));
        const f16* bp = W0 + ((size_t)(kc * 8 + wave) * 512) + lane * 16;
        v16h b = join8(*(const v8h*)bp, *(const v8h*)(bp + 8));
        acc = __builtin_amdgcn_wmma_f32_16x16x32_f16(
            false, a, false, b, (short)0, acc, false, false);
    }

    const int f = wave * 16 + ii;
#pragma unroll
    for (int v = 0; v < 8; ++v) {
        int r = i0 + hf * 8 + v;
        f16 o = (f16)softplus_f(acc[v]);
        y0[(size_t)r * 128 + f] = o;
        if (r < 63) y0[(size_t)(NROWS + r) * 128 + f] = o;   // replicated tail
    }
}

// ---------------------------------------------------------------------------
// Conv layer: y'[i,f] = softplus( sum_{m,c} W[m*128+c,f] * y[(m-i)&mask, c] + b[f] )
// 32 output rows / workgroup. Inputs = 63 contiguous rows of the padded y
// buffer starting at rbase = (N - i0 - 31) & mask  (tail replication => no wrap).
// A[tile0: ii][m*128+c] = Y[m+31-ii][c];  A[tile1: 16+ii] = Y[m+15-ii][c].
// ---------------------------------------------------------------------------
__global__ __launch_bounds__(256) void conv_kernel(const f16* __restrict__ yprev,
                                                   const f16* __restrict__ W,
                                                   const float* __restrict__ bias,
                                                   f16* __restrict__ ynext) {
    __shared__ f16 Y[63 * 136];        // 63 rows x (256B + 16B pad) = 17136 B
    const int tid = threadIdx.x;
    const int i0  = blockIdx.x * 32;
    const int rbase = (NROWS - i0 - 31) & NMASK;

#if defined(__gfx1250__) && __has_builtin(__builtin_amdgcn_tensor_load_to_lds)
    if (tid < 32) {
        // Tensor DMA descriptor (CDNA5 ISA):
        // 2-D tile: 32 x 63 elements of 8B (= 63 rows x 256B), row stride 32,
        // LDS padding 4 DWORDs every 64 DWORDs (pad_interval=5, pad_amount=3).
        unsigned long long ga = (unsigned long long)(yprev + (size_t)rbase * 128);
        u32x4 g0;
        g0[0] = 1u;                                    // count=1 (valid)
        g0[1] = 0u;                                    // lds_addr: Y at LDS base
        g0[2] = (unsigned)(ga & 0xFFFFFFFFu);
        g0[3] = (unsigned)((ga >> 32) & 0x01FFFFFFu) | 0x80000000u;  // type=2
        i32x8 g1;
        g1[0] = (3 << 16) | (1 << 20) | (5 << 22) | (3 << 25);
        g1[1] = (int)(32u << 16);                      // tensor_dim0 = 32
        g1[2] = (int)(63u << 16);                      // tensor_dim1 = 63
        g1[3] = (int)(32u << 16);                      // tile_dim0  = 32
        g1[4] = 63;                                    // tile_dim1  = 63
        g1[5] = 32;                                    // tensor_dim0_stride = 32
        g1[6] = 0;
        g1[7] = 0;
        i32x4 z4 = {0, 0, 0, 0};
#if defined(__clang_major__) && (__clang_major__ >= 23)
        i32x8 z8 = {0, 0, 0, 0, 0, 0, 0, 0};
        __builtin_amdgcn_tensor_load_to_lds(g0, g1, z4, z4, z8, 0);
#else
        __builtin_amdgcn_tensor_load_to_lds(g0, g1, z4, z4, 0);
#endif
        __builtin_amdgcn_s_wait_tensorcnt(0);
    }
    __syncthreads();
#else
    for (int idx = tid; idx < 63 * 16; idx += 256) {
        int t = idx >> 4, seg = idx & 15;
        *(v8h*)&Y[t * 136 + seg * 8] =
            *(const v8h*)(yprev + (size_t)(rbase + t) * 128 + seg * 8);
    }
    __syncthreads();
#endif

    const int wave = tid >> 5, lane = tid & 31;
    const int ii = lane & 15, hf = lane >> 4;

    v8f acc0 = {}, acc1 = {};
#pragma unroll 4
    for (int kc = 0; kc < 128; ++kc) {
        int m = kc >> 2, c0 = (kc & 3) * 32;
        const f16* bp = W + ((size_t)(kc * 8 + wave) * 512) + lane * 16;
        __builtin_prefetch(bp + 8 * 4096, 0, 1);       // B tile prefetch, 8 K-chunks ahead
        v16h b = join8(*(const v8h*)bp, *(const v8h*)(bp + 8));
        const f16* ap0 = &Y[(m + 31 - ii) * 136 + c0 + hf * 8];
        v16h a0 = join8(*(const v8h*)ap0, *(const v8h*)(ap0 + 16));
        acc0 = __builtin_amdgcn_wmma_f32_16x16x32_f16(
            false, a0, false, b, (short)0, acc0, false, false);
        const f16* ap1 = &Y[(m + 15 - ii) * 136 + c0 + hf * 8];
        v16h a1 = join8(*(const v8h*)ap1, *(const v8h*)(ap1 + 16));
        acc1 = __builtin_amdgcn_wmma_f32_16x16x32_f16(
            false, a1, false, b, (short)0, acc1, false, false);
    }

    const int f = wave * 16 + ii;
    const float bv = bias[f];
#pragma unroll
    for (int v = 0; v < 8; ++v) {
        int r0 = i0 + hf * 8 + v;
        int r1 = r0 + 16;
        f16 o0 = (f16)softplus_f(acc0[v] + bv);
        f16 o1 = (f16)softplus_f(acc1[v] + bv);
        ynext[(size_t)r0 * 128 + f] = o0;
        ynext[(size_t)r1 * 128 + f] = o1;
        if (r0 < 63) ynext[(size_t)(NROWS + r0) * 128 + f] = o0;
        if (r1 < 63) ynext[(size_t)(NROWS + r1) * 128 + f] = o1;
    }
}

// ---------------------------------------------------------------------------
// Final reduction: out = (1/N) * sum_{i,c} y2[i,c] * w[c] + dense_b
// ---------------------------------------------------------------------------
__global__ __launch_bounds__(256) void reduce1(const f16* __restrict__ y2,
                                               const float* __restrict__ w,
                                               float* __restrict__ partials) {
    __shared__ float sm[256];
    const int tid = threadIdx.x;
    float acc = 0.f;
    const size_t total = (size_t)NROWS * 128;
    for (size_t idx = (size_t)blockIdx.x * 256 + tid; idx < total;
         idx += (size_t)gridDim.x * 256) {
        acc += (float)y2[idx] * w[idx & 127];
    }
    sm[tid] = acc;
    __syncthreads();
    for (int s = 128; s > 0; s >>= 1) {
        if (tid < s) sm[tid] += sm[tid + s];
        __syncthreads();
    }
    if (tid == 0) partials[blockIdx.x] = sm[0];
}

__global__ __launch_bounds__(256) void reduce2(const float* __restrict__ partials,
                                               int np,
                                               const float* __restrict__ dense_b,
                                               float* __restrict__ out) {
    __shared__ float sm[256];
    float acc = 0.f;
    for (int i = threadIdx.x; i < np; i += 256) acc += partials[i];
    sm[threadIdx.x] = acc;
    __syncthreads();
    for (int s = 128; s > 0; s >>= 1) {
        if (threadIdx.x < s) sm[threadIdx.x] += sm[threadIdx.x + s];
        __syncthreads();
    }
    if (threadIdx.x == 0) out[0] = sm[0] / (float)NROWS + dense_b[0];
}

// ---------------------------------------------------------------------------
extern "C" void kernel_launch(void* const* d_in, const int* in_sizes, int n_in,
                              void* d_out, int out_size, void* d_ws, size_t ws_size,
                              hipStream_t stream) {
    const int*   x  = (const int*)  d_in[0];
    const float* ik = (const float*)d_in[1];   // input_kernel (128,32,32)
    const float* k0 = (const float*)d_in[2];   // kernel_0 (32,128,128)
    const float* b0 = (const float*)d_in[3];   // bias_0 (128)
    const float* k1 = (const float*)d_in[4];   // kernel_1 (32,128,128)
    const float* b1 = (const float*)d_in[5];   // bias_1 (128)
    const float* dw = (const float*)d_in[6];   // dense_w (128)
    const float* db = (const float*)d_in[7];   // dense_b (1)

    char* ws = (char*)d_ws;
    const size_t ybytes = (size_t)(NROWS + YPAD) * 128 * 2;   // padded y buffer
    size_t off = 0;
    f16* W0 = (f16*)(ws + off); off += (size_t)1024 * 128 * 2;
    f16* W1 = (f16*)(ws + off); off += (size_t)4096 * 128 * 2;
    f16* W2 = (f16*)(ws + off); off += (size_t)4096 * 128 * 2;
    f16* y0 = (f16*)(ws + off); off += ybytes;
    f16* y1 = (f16*)(ws + off); off += ybytes;
    f16* y2 = (f16*)(ws + off); off += ybytes;
    float* partials = (float*)(ws + off);

    prep_weights<<<4608, 256, 0, stream>>>(ik, k0, k1, W0, W1, W2);
    layer0_kernel<<<NROWS / 16, 256, 0, stream>>>(x, W0, y0);
    conv_kernel<<<NROWS / 32, 256, 0, stream>>>(y0, W1, b0, y1);
    conv_kernel<<<NROWS / 32, 256, 0, stream>>>(y1, W2, b1, y2);
    reduce1<<<256, 256, 0, stream>>>(y2, dw, partials);
    reduce2<<<1, 256, 0, stream>>>(partials, 256, db, (float*)d_out);
}
